// MultiHeadedAttention_42958262894841
// MI455X (gfx1250) — matrix operand order repaired and checked by reading, not compiled, not run on hardware
//
#include <hip/hip_runtime.h>
#include <stdint.h>

// ---------------------------------------------------------------------------
// MI455X (gfx1250) fused attention:
//   transpose_v   : V f32 [B,S,DV] -> Vt bf16 [B,DV,S]   (for contiguous PV loads)
//   proj_gemm     : Q/K projections, WMMA bf16, TDM double-buffered LDS tiles
//   flash_attn    : fused QK^T -> online softmax -> PV, all WMMA bf16
// Shapes: B=8, S=2048, D_MODEL=1024, D_K=D_V=512.
// ---------------------------------------------------------------------------

typedef __attribute__((ext_vector_type(16))) __bf16 v16bf;
typedef __attribute__((ext_vector_type(8)))  __bf16 v8bf;
typedef __attribute__((ext_vector_type(8)))  float  v8f;

#define B_SZ     8
#define S_LEN    2048
#define D_MODEL  1024
#define DK       512
#define DV       512
#define KEY_TILE 128
#define PJ_N     128                 // proj block tile: 16 (M) x 128 (N)
#define NEG_INF  (-1.0e9f)

// ---------------------------------------------------------------------------
// Tensor Data Mover support (6-arg clang-23 builtin; ISA 08_async_tensor D#)
// ---------------------------------------------------------------------------
#if defined(__has_builtin)
#  if __has_builtin(__builtin_amdgcn_tensor_load_to_lds) && \
      __has_builtin(__builtin_amdgcn_s_wait_tensorcnt)
#    define HAVE_TDM 1
#  else
#    define HAVE_TDM 0
#  endif
#else
#  define HAVE_TDM 0
#endif

#if HAVE_TDM
typedef unsigned int tdm_u32x4 __attribute__((ext_vector_type(4)));
typedef int          tdm_i32x8 __attribute__((ext_vector_type(8)));
typedef int          tdm_i32x4 __attribute__((ext_vector_type(4)));

// 2-D tile load: tile (tile0 x tile1) elements from a row-major tensor whose
// dim0 line stride is stride0 elements.  elem_code: 0=1B,1=2B,2=4B.
__device__ __forceinline__ void tdm_load_2d(uint32_t lds_off, const void* gptr,
                                            uint32_t elem_code,
                                            uint32_t td0, uint32_t td1,
                                            uint32_t tile0, uint32_t tile1,
                                            uint64_t stride0) {
  const uint64_t ga = (uint64_t)(uintptr_t)gptr;
  tdm_u32x4 g0;
  g0[0] = 1u;                                             // count=1, user D#
  g0[1] = lds_off;                                        // lds_addr [63:32]
  g0[2] = (uint32_t)ga;                                   // global_addr lo
  g0[3] = (uint32_t)((ga >> 32) & 0x01FFFFFFu) | (2u << 30); // addr hi | type=2
  tdm_i32x8 g1;
  g1[0] = (int)(elem_code << 16);                         // wg_mask=0, data_size
  g1[1] = (int)((td0 & 0xFFFFu) << 16);                   // tensor_dim0 [79:48]
  g1[2] = (int)((td0 >> 16) | ((td1 & 0xFFFFu) << 16));   // tensor_dim1 [111:80]
  g1[3] = (int)((td1 >> 16) | ((tile0 & 0xFFFFu) << 16)); // tile_dim0 [127:112]
  g1[4] = (int)(tile1 & 0xFFFFu);                         // tile_dim1, tile_dim2=0
  g1[5] = (int)(uint32_t)stride0;                         // dim0_stride lo
  g1[6] = (int)((uint32_t)(stride0 >> 32) & 0xFFFFu);     // dim0_stride hi
  g1[7] = 0;
  tdm_i32x4 z4 = {0, 0, 0, 0};                            // 2-D: groups 2/3 unused
  tdm_i32x8 z8 = {0, 0, 0, 0, 0, 0, 0, 0};
  __builtin_amdgcn_tensor_load_to_lds(g0, g1, z4, z4, z8, 0);
}
#endif

// ---------------------------------------------------------------------------
// V [B,S,DV] f32  ->  Vt [B,DV,S] bf16  (tiled transpose through LDS)
// ---------------------------------------------------------------------------
__global__ __launch_bounds__(256) void transpose_v_bf16(const float* __restrict__ V,
                                                        __bf16* __restrict__ Vt) {
  __shared__ float tile[32][33];
  const int b  = blockIdx.z;
  const int s0 = blockIdx.x * 32;
  const int d0 = blockIdx.y * 32;
  const int tx = threadIdx.x & 31, ty = threadIdx.x >> 5;   // 32 x 8
#pragma unroll
  for (int i = ty; i < 32; i += 8)
    tile[i][tx] = V[((size_t)b * S_LEN + s0 + i) * DV + d0 + tx];
  __syncthreads();
#pragma unroll
  for (int i = ty; i < 32; i += 8)
    Vt[((size_t)b * DV + d0 + i) * S_LEN + s0 + tx] = (__bf16)tile[tx][i];
}

// ---------------------------------------------------------------------------
// C[M,512] = X[M,1024] @ W[1024,512] + bias, output bf16.
// Block tile 16 (M) x 128 (N); 8 waves each own a 16x16 column tile.
// A/W k-tiles staged in LDS; TDM double-buffered when available.
// ---------------------------------------------------------------------------
__global__ __launch_bounds__(256) void proj_gemm_bf16(const float* __restrict__ X,
                                                      const float* __restrict__ W,
                                                      const float* __restrict__ bias,
                                                      __bf16* __restrict__ Out,
                                                      int M) {
  __shared__ float s_a[2][16][32];     // 4 KB
  __shared__ float s_w[2][32][PJ_N];   // 32 KB

  const int tid  = threadIdx.x;
  const int lane = tid & 31, wave = tid >> 5;
  const int nBlk = DK / PJ_N;                       // 4
  const int row0 = (blockIdx.x / nBlk) * 16;
  const int n0   = (blockIdx.x % nBlk) * PJ_N;
  if (row0 >= M) return;                            // block-uniform
  const int ml   = lane & 15, half = lane >> 4;
  const int cl   = wave * 16 + ml;                  // column within block tile

#if HAVE_TDM
  const uint32_t ldsA[2] = { (uint32_t)(uintptr_t)&s_a[0][0][0],
                             (uint32_t)(uintptr_t)&s_a[1][0][0] };
  const uint32_t ldsW[2] = { (uint32_t)(uintptr_t)&s_w[0][0][0],
                             (uint32_t)(uintptr_t)&s_w[1][0][0] };
  if (wave == 0) {  // one wave drives the DMA engine (EXEC ignored by TDM)
    tdm_load_2d(ldsA[0], X + (size_t)row0 * D_MODEL, 2,
                D_MODEL, (uint32_t)M, 32, 16, D_MODEL);
    tdm_load_2d(ldsW[0], W + n0, 2, DK, D_MODEL, PJ_N, 32, DK);
  }
#endif

  v8f acc = {};
  int buf = 0;
  for (int kk = 0; kk < D_MODEL; kk += 32, buf ^= 1) {
#if HAVE_TDM
    if (wave == 0) {
      if (kk + 32 < D_MODEL) {
        // issue next k-tile into the other buffer, then wait for current pair
        tdm_load_2d(ldsA[buf ^ 1], X + (size_t)row0 * D_MODEL + kk + 32, 2,
                    D_MODEL, (uint32_t)M, 32, 16, D_MODEL);
        tdm_load_2d(ldsW[buf ^ 1], W + (size_t)(kk + 32) * DK + n0, 2,
                    DK, D_MODEL, PJ_N, 32, DK);
        __builtin_amdgcn_s_wait_tensorcnt(2);   // current buffer's pair done
      } else {
        __builtin_amdgcn_s_wait_tensorcnt(0);
      }
    }
#else
    {  // cooperative synchronous staging fallback
      const int i0 = tid * 2, i1 = i0 + 1;
      s_a[buf][i0 >> 5][i0 & 31] =
          X[(size_t)(row0 + (i0 >> 5)) * D_MODEL + kk + (i0 & 31)];
      s_a[buf][i1 >> 5][i1 & 31] =
          X[(size_t)(row0 + (i1 >> 5)) * D_MODEL + kk + (i1 & 31)];
#pragma unroll
      for (int i = 0; i < 16; ++i) {
        const int idx = tid + i * 256, r = idx >> 7, c = idx & 127;
        s_w[buf][r][c] = W[(size_t)(kk + r) * DK + n0 + c];
      }
    }
#endif
    __syncthreads();

    // A fragment (shared by all 8 waves): lane=row ml;
    // elems 0..7 = K(8*half+i), 8..15 = K(16+8*half+i)
    v16bf a, bfr;
    const float* ar = &s_a[buf][ml][8 * half];
#pragma unroll
    for (int i = 0; i < 8; ++i) {
      a[i]     = (__bf16)ar[i];
      a[8 + i] = (__bf16)ar[16 + i];
    }
    // B fragment: lane=col cl; elems 0..7 = K(8*half+i), 8..15 = K(16+8*half+i)
#pragma unroll
    for (int i = 0; i < 16; ++i) bfr[i] = (__bf16)s_w[buf][16 * (i >> 3) + 8 * half + (i & 7)][cl];

    acc = __builtin_amdgcn_wmma_f32_16x16x32_bf16(false, a, false, bfr,
                                                  (short)0, acc, false, false);
    __syncthreads();   // everyone done with buf before TDM refills it
  }

  const float bv = bias[n0 + cl];
#pragma unroll
  for (int r = 0; r < 8; ++r)
    Out[(size_t)(row0 + r + 8 * half) * DK + n0 + cl] = (__bf16)(acc[r] + bv);
}

// ---------------------------------------------------------------------------
// Fused flash attention.  grid = B*(S/16), 256 threads (8 waves).
// Per 128-key tile: wave w scores keys [w*16,w*16+16) (16 WMMAs, K=512),
// cooperative online softmax in LDS, wave w accumulates P@V for d_v slice
// [w*64,(w+1)*64) (16 WMMAs) reading Vt (contiguous b128 loads).
// ---------------------------------------------------------------------------
__global__ __launch_bounds__(256) void flash_attn(const __bf16* __restrict__ Qg,
                                                  const __bf16* __restrict__ Kg,
                                                  const __bf16* __restrict__ Vt,
                                                  const int*    __restrict__ maskg,
                                                  float*        __restrict__ Out) {
  __shared__ float  s_scores[16][KEY_TILE];
  __shared__ __bf16 s_p[16][KEY_TILE];
  __shared__ float  s_red[16][16];
  __shared__ float  s_alpha[16];
  __shared__ float  s_runmax[16];
  __shared__ float  s_runsum[16];

  const int qTiles = S_LEN / 16;
  const int bIdx   = blockIdx.x / qTiles;
  const int row0   = (blockIdx.x % qTiles) * 16;

  const int tid  = threadIdx.x;
  const int lane = tid & 31;
  const int wave = tid >> 5;
  const int nl   = lane & 15;
  const int half = lane >> 4;

  const float scale = 0.044194173824159216f;   // 1/sqrt(512)

  if (tid < 16) { s_runmax[tid] = NEG_INF; s_runsum[tid] = 0.0f; }
  __syncthreads();

  // preload Q A-fragments (K-dim 512 = 16 fragments, kept in VGPRs)
  v16bf qa[16];
  {
    const __bf16* qrow = Qg + ((size_t)bIdx * S_LEN + row0 + nl) * DK;
#pragma unroll
    for (int c = 0; c < 16; ++c) {
      qa[c] = *(const v16bf*)(qrow + c * 32 + 16 * half);
    }
  }

  v8f accv[4] = {};
  const int dvBase = wave * 64;

  for (int key0 = 0; key0 < S_LEN; key0 += KEY_TILE) {
    // ---- 1. scores for this wave's 16-key slice ---------------------------
    const int kslice = key0 + wave * 16;
    {
      v8f sc = {};
      const __bf16* krow = Kg + ((size_t)bIdx * S_LEN + kslice + nl) * DK;
#pragma unroll
      for (int c = 0; c < 16; ++c) {
        v16bf kb = *(const v16bf*)(krow + c * 32 + 16 * half);
        sc = __builtin_amdgcn_wmma_f32_16x16x32_bf16(false, qa[c], false, kb,
                                                     (short)0, sc, false, false);
      }
      if (key0 + KEY_TILE < S_LEN) {
        __builtin_prefetch(krow + (size_t)KEY_TILE * DK, 0, 1);   // next K tile
        __builtin_prefetch(Vt + ((size_t)bIdx * DV + dvBase + nl) * S_LEN +
                               key0 + KEY_TILE, 0, 1);            // next V tile
      }
      const bool masked = (maskg[(size_t)bIdx * S_LEN + kslice + nl] == 0);
#pragma unroll
      for (int r = 0; r < 8; ++r)
        s_scores[r + 8 * half][wave * 16 + nl] = masked ? NEG_INF : sc[r] * scale;
    }
    __syncthreads();

    // ---- 2. per-row running max / alpha -----------------------------------
    {
      const int row = tid >> 4, j = tid & 15;
      float m = NEG_INF;
#pragma unroll
      for (int i = 0; i < 8; ++i) m = fmaxf(m, s_scores[row][j * 8 + i]);
      s_red[row][j] = m;
    }
    __syncthreads();
    if (tid < 16) {
      float tmax = NEG_INF;
#pragma unroll
      for (int j = 0; j < 16; ++j) tmax = fmaxf(tmax, s_red[tid][j]);
      const float newmax = fmaxf(s_runmax[tid], tmax);
      s_alpha[tid]  = __expf(s_runmax[tid] - newmax);
      s_runmax[tid] = newmax;
    }
    __syncthreads();

    // ---- 3. P = exp(S - max), running sum ---------------------------------
    {
      const int row = tid >> 4, j = tid & 15;
      const float rmax = s_runmax[row];
      float sum = 0.0f;
#pragma unroll
      for (int i = 0; i < 8; ++i) {
        const float p = __expf(s_scores[row][j * 8 + i] - rmax);
        s_p[row][j * 8 + i] = (__bf16)p;
        sum += p;
      }
      s_red[row][j] = sum;
    }
    __syncthreads();
    if (tid < 16) {
      float tsum = 0.0f;
#pragma unroll
      for (int j = 0; j < 16; ++j) tsum += s_red[tid][j];
      s_runsum[tid] = s_runsum[tid] * s_alpha[tid] + tsum;
    }
    __syncthreads();

    // ---- 4. rescale accumulators, P @ V (contiguous Vt loads) -------------
#pragma unroll
    for (int ct = 0; ct < 4; ++ct)
#pragma unroll
      for (int r = 0; r < 8; ++r) accv[ct][r] *= s_alpha[r + 8 * half];

#pragma unroll
    for (int ct = 0; ct < 4; ++ct) {
      const int dv0 = dvBase + ct * 16;
      // B fragment: lane = d_v column (dv0+nl); elem i = key (c2*32+16*half+i)
      const __bf16* vrow = Vt + ((size_t)bIdx * DV + dv0 + nl) * S_LEN + key0;
#pragma unroll
      for (int c2 = 0; c2 < 4; ++c2) {
        v16bf pa = *(const v16bf*)&s_p[nl][c2 * 32 + 16 * half];
        v16bf vb = *(const v16bf*)(vrow + c2 * 32 + 16 * half);
        accv[ct] = __builtin_amdgcn_wmma_f32_16x16x32_bf16(false, pa, false, vb,
                                                           (short)0, accv[ct],
                                                           false, false);
      }
    }
    __syncthreads();
  }

  // ---- epilogue ----------------------------------------------------------
#pragma unroll
  for (int r = 0; r < 8; ++r) {
    const int qrow = r + 8 * half;
    const float rcp = 1.0f / s_runsum[qrow];
#pragma unroll
    for (int ct = 0; ct < 4; ++ct) {
      Out[((size_t)bIdx * S_LEN + row0 + qrow) * DV + dvBase + ct * 16 + nl] =
          accv[ct][r] * rcp;
    }
  }
}

// ---------------------------------------------------------------------------
// launch
// ---------------------------------------------------------------------------
extern "C" void kernel_launch(void* const* d_in, const int* in_sizes, int n_in,
                              void* d_out, int out_size, void* d_ws, size_t ws_size,
                              hipStream_t stream) {
  const float* query = (const float*)d_in[0];
  const float* key_  = (const float*)d_in[1];
  const float* value = (const float*)d_in[2];
  const int*   mask  = (const int*)d_in[3];
  const float* Wq    = (const float*)d_in[4];
  const float* bq    = (const float*)d_in[5];
  const float* Wk    = (const float*)d_in[6];
  const float* bk    = (const float*)d_in[7];
  float* out = (float*)d_out;

  const size_t elems = (size_t)B_SZ * S_LEN * DK;
  __bf16* Qb = (__bf16*)d_ws;
  __bf16* Kb = Qb + elems;
  __bf16* Vt = Kb + elems;

  const int M = B_SZ * S_LEN;   // 16384

  // V -> bf16, transposed to [B, DV, S]
  transpose_v_bf16<<<dim3(S_LEN / 32, DV / 32, B_SZ), dim3(256), 0, stream>>>(value, Vt);

  // Q and K projections
  {
    const int blocks = (M / 16) * (DK / PJ_N);   // 4096
    proj_gemm_bf16<<<dim3(blocks), dim3(256), 0, stream>>>(query, Wq, bq, Qb, M);
    proj_gemm_bf16<<<dim3(blocks), dim3(256), 0, stream>>>(key_,  Wk, bk, Kb, M);
  }

  // fused flash attention
  flash_attn<<<dim3(B_SZ * (S_LEN / 16)), dim3(256), 0, stream>>>(Qb, Kb, Vt, mask, out);
}
